// NT_Xent_10204842296042
// MI455X (gfx1250) — matrix operand order repaired and checked by reading, not compiled, not run on hardware
//
#include <hip/hip_runtime.h>
#include <hip/hip_bf16.h>
#include <stdint.h>

// ---------------------------------------------------------------------------
// NT-Xent (SimCLR) fused loss for B=4096, D=512 on gfx1250 (MI455X).
//
// Compute-bound: 8192x8192x512 MAC GEMM (6.9e10 FLOPs) vs 16.8 MB input.
// Strategy: row-normalize to f16 (unit vectors -> f16 is safe), fused
// v_wmma_f32_16x16x32_f16 GEMM with streaming exp/row-sum so the 268 MB
// similarity matrix is never materialized. f32 accumulation throughout.
//
// CDNA5-specific paths used:
//   * v_wmma_f32_16x16x32_f16 (XDL matrix path, wave32 fragments)
//   * global_load_async_to_lds_b128 + s_wait_asynccnt (A-panel staging,
//     VGPR-bypassing async DMA into padded LDS layout)
//   * ds_load_b128 fragment feeds, split-counter waits, wave32 shuffles
// ---------------------------------------------------------------------------

typedef __attribute__((ext_vector_type(16))) _Float16 v16h;
typedef __attribute__((ext_vector_type(8)))  _Float16 v8h;
typedef __attribute__((ext_vector_type(8)))  float    v8f;

#define NROWS   8192      // 2B
#define D       512
#define INV_T   2.0f      // 1 / 0.5

// LDS strides (halves), padded for conflict-free ds_load_b128 fragment reads
#define A_STRIDE 72       // 64 K + 8 pad  (144 B, 16B aligned)
#define B_STRIDE 136      // 128 N + 8 pad (272 B, 16B aligned)

union V16U { v16h v; v8h h[2]; };

// ---------------------------------------------------------------------------
// Kernel 1: L2-normalize rows of concat(z_i, z_j) -> f16 matrix Zh[8192][512]
// One wave per row; lane covers 16 contiguous columns via 4 float4 loads.
// ---------------------------------------------------------------------------
__global__ __launch_bounds__(256) void ntxent_normalize(
    const float* __restrict__ zi, const float* __restrict__ zj,
    _Float16* __restrict__ Zh)
{
    const int wave = threadIdx.x >> 5;
    const int lane = threadIdx.x & 31;
    const int row  = blockIdx.x * 8 + wave;

    const float* src = (row < 4096) ? (zi + (size_t)row * D)
                                    : (zj + (size_t)(row - 4096) * D);
    const float4* p = (const float4*)src;

    float4 v[4];
    float ss = 0.0f;
#pragma unroll
    for (int j = 0; j < 4; ++j) {
        v[j] = p[lane * 4 + j];
        ss += v[j].x * v[j].x + v[j].y * v[j].y + v[j].z * v[j].z + v[j].w * v[j].w;
    }
#pragma unroll
    for (int m = 1; m < 32; m <<= 1) ss += __shfl_xor(ss, m, 32);
    const float inv = rsqrtf(ss);

    _Float16* out = Zh + (size_t)row * D + lane * 16;
#pragma unroll
    for (int j = 0; j < 4; ++j) {
        out[j * 4 + 0] = (_Float16)(v[j].x * inv);
        out[j * 4 + 1] = (_Float16)(v[j].y * inv);
        out[j * 4 + 2] = (_Float16)(v[j].z * inv);
        out[j * 4 + 3] = (_Float16)(v[j].w * inv);
    }
}

// ---------------------------------------------------------------------------
// Kernel 2: zero per-row partial buffers (rsum[8192], pos[8192])
// ---------------------------------------------------------------------------
__global__ void ntxent_init(float* __restrict__ buf)
{
    buf[blockIdx.x * 256 + threadIdx.x] = 0.0f;   // 64 * 256 = 16384 floats
}

// ---------------------------------------------------------------------------
// Kernel 3: fused WMMA GEMM + exp row-sum + positive-pair capture.
//
// Grid = 1024 blocks = 64 row-panels (128 rows) x 16 col-groups (512 cols).
// Block = 256 threads = 8 waves; wave w owns rows [panel + 16w, +16).
// Per 128-col block each wave keeps 8 v8f accumulator tiles, K accumulated
// in 64-wide LDS-staged chunks (2 wmma k-steps per chunk).
//
// Fragment addressing follows CDNA5 ISA 7.12.2 (wave32):
//   A 16x32 f16: lane m<16 -> row m, K {0..7,16..23}; lane m>=16 -> +8 K off.
//   B 32x16 f16: lane n<16 -> col n, same K split as A; staged transposed in LDS so the
//                16 halves per lane are 16 separate 16-bit loads, one per k-row.
//   C/D:         VGPR v -> row v (lanes 0-15) / v+8 (lanes 16-31), N = lane%16.
// ---------------------------------------------------------------------------
__global__ __launch_bounds__(256) void ntxent_main(
    const _Float16* __restrict__ Zh,
    float* __restrict__ rsum, float* __restrict__ pos)
{
    __shared__ _Float16 Asm[128 * A_STRIDE];   // 18,432 B : 128 rows x 64 K
    __shared__ _Float16 Bsm[64  * B_STRIDE];   // 17,408 B : 64 K x 128 N (transposed)

    const int tid  = threadIdx.x;
    const int wave = tid >> 5;
    const int lane = tid & 31;
    const int lo16 = lane & 15;
    const bool hi  = lane >= 16;

    const int rowPanel = (blockIdx.x >> 4) * 128;
    const int colGrp   = (blockIdx.x & 15) * 512;
    const int rowBase  = rowPanel + wave * 16;

    const uint64_t zhBase = (uint64_t)(uintptr_t)Zh;

    // Pre-computed per-thread staging indices
    const int aR = tid >> 1;                 // async A: row pair base (see below)
    const int cp = tid & 63;                 // B: column-pair index (cols 2cp, 2cp+1)
    const int kg = tid >> 6;                 // B: k-group of 16 (0..3)

    float rs[8], ps[8];
#pragma unroll
    for (int v = 0; v < 8; ++v) { rs[v] = 0.0f; ps[v] = 0.0f; }

    const v8f zero = {0.f, 0.f, 0.f, 0.f, 0.f, 0.f, 0.f, 0.f};

    for (int cb = 0; cb < 4; ++cb) {
        const int colBlock = colGrp + cb * 128;

        v8f acc[8];
#pragma unroll
        for (int ct = 0; ct < 8; ++ct) acc[ct] = zero;

        for (int kc = 0; kc < 8; ++kc) {
            const int k0 = kc * 64;

            // ---- stage A panel via async global->LDS DMA ----------------
            // 128 rows x 128 B (64 halves), dest stride 144 B. 1024 x 16 B
            // transfers = 4 per thread. Bypasses VGPRs, tracked by ASYNCcnt.
#pragma unroll
            for (int i = 0; i < 4; ++i) {
                const int idx = tid + i * 256;         // 0..1023
                const int r = idx >> 3, c = idx & 7;   // 8 x 16B chunks per row
                const uint32_t ldsAddr =
                    (uint32_t)(uintptr_t)(&Asm[r * A_STRIDE + c * 8]);
                const uint32_t gOff =
                    (uint32_t)(((rowPanel + r) * D + k0 + c * 8) * 2);
                asm volatile(
                    "global_load_async_to_lds_b128 %0, %1, %2"
                    :: "v"(ldsAddr), "v"(gOff), "s"(zhBase) : "memory");
            }

            // ---- stage B panel transposed: Bsm[k][n] = Zh[col][k] -------
            // Each thread: cols (2cp, 2cp+1), k rows kg*16..+15.
            // 4 global b128 loads, 16 ds_store_b32 (conflict-free: lanes of
            // a wave write 32 consecutive dwords of one k-row).
            {
                const _Float16* s0 =
                    Zh + (size_t)(colBlock + 2 * cp) * D + k0 + kg * 16;
                const _Float16* s1 = s0 + D;
                const v8h c0a = *(const v8h*)(s0);
                const v8h c0b = *(const v8h*)(s0 + 8);
                const v8h c1a = *(const v8h*)(s1);
                const v8h c1b = *(const v8h*)(s1 + 8);
                uint* dstB = (uint*)Bsm;               // 68 dwords per k-row
#pragma unroll
                for (int j = 0; j < 8; ++j) {
                    union { _Float16 h[2]; uint u; } pk;
                    pk.h[0] = c0a[j]; pk.h[1] = c1a[j];
                    dstB[(kg * 16 + j) * (B_STRIDE / 2) + cp] = pk.u;
                }
#pragma unroll
                for (int j = 0; j < 8; ++j) {
                    union { _Float16 h[2]; uint u; } pk;
                    pk.h[0] = c0b[j]; pk.h[1] = c1b[j];
                    dstB[(kg * 16 + 8 + j) * (B_STRIDE / 2) + cp] = pk.u;
                }
            }

            asm volatile("s_wait_asynccnt 0" ::: "memory");
            __syncthreads();

#pragma unroll
            for (int ks = 0; ks < 2; ++ks) {
                // A fragment: row = wave*16 + lane%16, K offset 0/8 per half
                const _Float16* ab =
                    &Asm[(wave * 16 + lo16) * A_STRIDE + ks * 32 + (hi ? 8 : 0)];
                V16U a;
                a.h[0] = *(const v8h*)(ab);
                a.h[1] = *(const v8h*)(ab + 16);

#pragma unroll
                for (int ct = 0; ct < 8; ++ct) {
                    // B fragment: N = lane%16, K {0..7,16..23} (+8 for lanes>=16), strided k-rows
                    const _Float16* bb = &Bsm[(ks * 32 + (hi ? 8 : 0)) * B_STRIDE + ct * 16 + lo16];
                    V16U b;
#pragma unroll
                    for (int j = 0; j < 8; ++j) {
                        b.v[j]     = bb[j * B_STRIDE];
                        b.v[8 + j] = bb[(16 + j) * B_STRIDE];
                    }

                    acc[ct] = __builtin_amdgcn_wmma_f32_16x16x32_f16(
                        false, a.v, false, b.v, (short)0, acc[ct], false, false);
                }
            }
            __syncthreads();
        }

        // ---- post-process this 16x128 strip: exp row-sums + positive logit
#pragma unroll
        for (int ct = 0; ct < 8; ++ct) {
            const int gc = colBlock + ct * 16 + lo16;
#pragma unroll
            for (int v = 0; v < 8; ++v) {
                const int gr = rowBase + v + (hi ? 8 : 0);
                const float s = acc[ct][v] * INV_T;
                if (gc != gr) rs[v] += __expf(s);
                const int partner = (gr < 4096) ? gr + 4096 : gr - 4096;
                if (gc == partner) ps[v] += s;   // log(exp(s)) = s
            }
        }
    }

    // ---- reduce across the 16 lanes that share each row, then atomics ----
#pragma unroll
    for (int v = 0; v < 8; ++v) {
        float r = rs[v], p = ps[v];
#pragma unroll
        for (int m = 1; m < 16; m <<= 1) {
            r += __shfl_xor(r, m, 32);
            p += __shfl_xor(p, m, 32);
        }
        if (lo16 == 0) {
            const int gr = rowBase + v + (hi ? 8 : 0);
            atomicAdd(&rsum[gr], r);
            atomicAdd(&pos[gr], p);
        }
    }
    (void)aR;
}

// ---------------------------------------------------------------------------
// Kernel 4: loss = -mean(pos_logit - log(den)) over 8192 rows
// ---------------------------------------------------------------------------
__global__ __launch_bounds__(256) void ntxent_finalize(
    const float* __restrict__ rsum, const float* __restrict__ pos,
    float* __restrict__ out)
{
    __shared__ float sdata[256];
    float acc = 0.0f;
    for (int r = threadIdx.x; r < NROWS; r += 256)
        acc += pos[r] - logf(rsum[r]);
    sdata[threadIdx.x] = acc;
    __syncthreads();
    for (int s = 128; s > 0; s >>= 1) {
        if (threadIdx.x < s) sdata[threadIdx.x] += sdata[threadIdx.x + s];
        __syncthreads();
    }
    if (threadIdx.x == 0) out[0] = -sdata[0] / (float)NROWS;
}

// ---------------------------------------------------------------------------
extern "C" void kernel_launch(void* const* d_in, const int* in_sizes, int n_in,
                              void* d_out, int out_size, void* d_ws, size_t ws_size,
                              hipStream_t stream)
{
    const float* zi = (const float*)d_in[0];
    const float* zj = (const float*)d_in[1];
    float* out = (float*)d_out;

    // workspace layout
    char* ws = (char*)d_ws;
    _Float16* Zh   = (_Float16*)ws;                        // 8192*512*2 = 8,388,608 B
    float*    rsum = (float*)(ws + (size_t)NROWS * D * 2); // 8192 floats
    float*    pos  = rsum + NROWS;                         // 8192 floats

    ntxent_normalize<<<NROWS / 8, 256, 0, stream>>>(zi, zj, Zh);
    ntxent_init<<<64, 256, 0, stream>>>(rsum);             // zeros rsum+pos (16384 floats)
    ntxent_main<<<1024, 256, 0, stream>>>(Zh, rsum, pos);
    ntxent_finalize<<<1, 256, 0, stream>>>(rsum, pos, out);
}
